// GraphContrastiveNetwork_5111011083069
// MI455X (gfx1250) — matrix operand order repaired and checked by reading, not compiled, not run on hardware
//
#include <hip/hip_runtime.h>
#include <hip/hip_bf16.h>
#include <math.h>

// ---------------------------------------------------------------------------
// GATConv (H=4, D=128) + linear head for MI455X (gfx1250, wave32, WMMA).
// GEMMs: v_wmma_f32_16x16x32_bf16, double-buffered LDS tiles, b128 global
// staging + packed bf16 b64 LDS stores. Scatter phase is the bandwidth bound
// (~3.2 GB @ 23.3 TB/s): float4 gathers + global_atomic_add_f32.
// ---------------------------------------------------------------------------

typedef __attribute__((ext_vector_type(16))) __bf16 v16bf;
typedef __attribute__((ext_vector_type(8)))  float  v8f;

namespace {
constexpr long kN    = 256L * 512L;   // 131072 nodes
constexpr int  kC    = 128;           // in channels
constexpr int  kH    = 4;             // heads
constexpr int  kD    = 128;           // out channels per head
constexpr int  kHD   = kH * kD;       // 512
constexpr long kE    = kN * 5;        // 655360 edges
constexpr long kEtot = kE + kN;       // + self loops = 786432
constexpr float kSlope = 0.2f;

constexpr int BM = 128, BN = 128, BK = 32;   // WMMA macro-tile
}

union Frag { v16bf v; unsigned int u[8]; };

// Per-wave 16x128 strip of a 128x128 tile: 8 x v_wmma_f32_16x16x32_bf16.
// A staged row-major [m][k] (bf16), B staged transposed [n][k] (bf16), so a
// packed K-pair is one aligned 32-bit LDS load (ds_load_b128 after merge).
__device__ inline void wmma_tile(const __bf16* sA, const __bf16* sB,
                                 v8f (&acc)[8], int lane, int wave) {
  const int half = lane >> 4, l16 = lane & 15, m0 = wave * 16;
  Frag a;
#pragma unroll
  for (int j = 0; j < 8; ++j) {
    const int ka = ((j < 4) ? 2 * j : 16 + 2 * (j - 4)) + half * 8;
    a.u[j] = *(const unsigned int*)(sA + (m0 + l16) * BK + ka);
  }
#pragma unroll
  for (int nn = 0; nn < 8; ++nn) {
    Frag b;
#pragma unroll
    for (int j = 0; j < 8; ++j) {
      const int kb = ((j < 4) ? 2 * j : 16 + 2 * (j - 4)) + half * 8;
      b.u[j] = *(const unsigned int*)(sB + (nn * 16 + l16) * BK + kb);
    }
    acc[nn] = __builtin_amdgcn_wmma_f32_16x16x32_bf16(
        false, a.v, false, b.v, (short)0, acc[nn], false, false);
  }
}

struct IdOp {
  __device__ float operator()(float v, int) const { return v; }
};
struct BiasEluOp {                       // fused x + bias_gat -> ELU
  const float* b;
  __device__ float operator()(float v, int k) const {
    const float t = v + b[k];
    return t > 0.f ? t : expm1f(t);
  }
};

// A tile: BM x BK fp32 -> bf16, b128 global loads, packed b64 LDS stores.
template <class F>
__device__ inline void stage_A(__bf16* sA, const float* __restrict__ g,
                               long row0, int ld, int k0, int tid, F f) {
#pragma unroll
  for (int j = 0; j < 4; ++j) {
    const int g4 = tid + j * 256;            // group of 4 elems, 1024 groups
    const int r  = g4 >> 3;                  // BK/4 == 8 groups per row
    const int c  = (g4 & 7) * 4;
    const float4 v = *(const float4*)&g[(row0 + r) * (long)ld + k0 + c];
    union { __bf16 h[4]; unsigned long long u; } pk;
    pk.h[0] = (__bf16)f(v.x, k0 + c + 0);
    pk.h[1] = (__bf16)f(v.y, k0 + c + 1);
    pk.h[2] = (__bf16)f(v.z, k0 + c + 2);
    pk.h[3] = (__bf16)f(v.w, k0 + c + 3);
    *(unsigned long long*)&sA[r * BK + c] = pk.u;   // ds_store_b64
  }
}

// B tile: BK x BN fp32, coalesced b128 loads along N, transposed [n][k] in LDS.
__device__ inline void stage_B(__bf16* sB, const float* __restrict__ g,
                               int ld, int k0, int col0, int tid) {
#pragma unroll
  for (int j = 0; j < 4; ++j) {
    const int g4 = tid + j * 256;
    const int c  = g4 >> 5;                  // BN/4 == 32 groups per k-row
    const int n  = (g4 & 31) * 4;
    const float4 v = *(const float4*)&g[(long)(k0 + c) * ld + col0 + n];
    sB[(n + 0) * BK + c] = (__bf16)v.x;
    sB[(n + 1) * BK + c] = (__bf16)v.y;
    sB[(n + 2) * BK + c] = (__bf16)v.z;
    sB[(n + 3) * BK + c] = (__bf16)v.w;
  }
}

// ---------------- GEMM1: xw[N,512] = x[N,128] @ W_gat[128,512] --------------
__global__ __launch_bounds__(256)
void gat_gemm1(const float* __restrict__ x, const float* __restrict__ W,
               float* __restrict__ xw) {
  __shared__ __align__(16) __bf16 sA[2][BM * BK];
  __shared__ __align__(16) __bf16 sB[2][BN * BK];
  const int tid  = threadIdx.x;
  const int bm   = blockIdx.x >> 2;       // kHD/BN == 4 column blocks
  const int bn   = blockIdx.x & 3;
  const long row0 = (long)bm * BM;
  const int  col0 = bn * BN;
  const int wave = tid >> 5, lane = tid & 31;
  v8f acc[8] = {};

  stage_A(sA[0], x, row0, kC, 0, tid, IdOp{});
  stage_B(sB[0], W, kHD, 0, col0, tid);
  __syncthreads();

  constexpr int nIter = kC / BK;          // 4
#pragma unroll
  for (int it = 0; it < nIter; ++it) {
    const int cur = it & 1;
    if (it + 1 < nIter) {                 // stage next tile into alt buffer
      stage_A(sA[cur ^ 1], x, row0, kC, (it + 1) * BK, tid, IdOp{});
      stage_B(sB[cur ^ 1], W, kHD, (it + 1) * BK, col0, tid);
    }
    wmma_tile(sA[cur], sB[cur], acc, lane, wave);
    __syncthreads();
  }

  const int half = lane >> 4, l16 = lane & 15, m0 = wave * 16;
#pragma unroll
  for (int nn = 0; nn < 8; ++nn)
#pragma unroll
    for (int r = 0; r < 8; ++r)
      xw[(row0 + m0 + r + 8 * half) * (long)kHD + col0 + nn * 16 + l16] =
          acc[nn][r];
}

// ------------- attention coefficients: a = <xw[n], att> per head ------------
__global__ __launch_bounds__(256)
void gat_attn(const float* __restrict__ xw, const float* __restrict__ att_s,
              const float* __restrict__ att_d, float* __restrict__ a_s,
              float* __restrict__ a_d) {
  const int wave = threadIdx.x >> 5, lane = threadIdx.x & 31;
  const long node = (long)blockIdx.x * 8 + wave;
  const float4* xr  = (const float4*)(xw + node * kHD) + lane * 4;
  const float4* as4 = (const float4*)att_s + lane * 4;   // att flat == channel
  const float4* ad4 = (const float4*)att_d + lane * 4;
  float ss = 0.f, sd = 0.f;
#pragma unroll
  for (int i = 0; i < 4; ++i) {
    const float4 xv = xr[i], av = as4[i], dv = ad4[i];
    ss += xv.x * av.x + xv.y * av.y + xv.z * av.z + xv.w * av.w;
    sd += xv.x * dv.x + xv.y * dv.y + xv.z * dv.z + xv.w * dv.w;
  }
#pragma unroll
  for (int o = 1; o < 8; o <<= 1) {       // reduce 8 lanes of the same head
    ss += __shfl_xor(ss, o, 32);
    sd += __shfl_xor(sd, o, 32);
  }
  if ((lane & 7) == 0) {
    a_s[node * kH + (lane >> 3)] = ss;
    a_d[node * kH + (lane >> 3)] = sd;
  }
}

// ------------------------------ init buffers --------------------------------
__global__ void gat_init(float4* __restrict__ acc4, float* __restrict__ m,
                         float* __restrict__ denom) {
  const long i = (long)blockIdx.x * blockDim.x + threadIdx.x;
  if (i < kN * kHD / 4) acc4[i] = make_float4(0.f, 0.f, 0.f, 0.f);
  if (i < kN * kH) { m[i] = -__builtin_inff(); denom[i] = 0.f; }
}

// monotone-int float max (m initialized to -inf)
__device__ inline void atomic_max_f32(float* addr, float val) {
  const int iv = __float_as_int(val);
  if (iv >= 0) atomicMax((int*)addr, iv);
  else         atomicMin((unsigned int*)addr, (unsigned int)iv);
}

// ------------------- pass 1: logits + segment max over dst ------------------
__global__ void gat_edge_logits(const long long* __restrict__ ei,
                                const float* __restrict__ a_s,
                                const float* __restrict__ a_d,
                                float* __restrict__ ebuf, float* __restrict__ m) {
  const long t = (long)blockIdx.x * blockDim.x + threadIdx.x;
  if (t >= kEtot * kH) return;
  const long e = t >> 2;
  const int  h = (int)(t & 3);
  long s, d;
  if (e < kE) { s = ei[e]; d = ei[kE + e]; } else { s = d = e - kE; }
  float v = a_s[s * kH + h] + a_d[d * kH + h];
  v = v > 0.f ? v : kSlope * v;            // leaky_relu
  ebuf[t] = v;
  atomic_max_f32(&m[d * kH + h], v);
}

// ---------------------- pass 2: exp + segment sum ---------------------------
__global__ void gat_edge_exp(const long long* __restrict__ ei,
                             float* __restrict__ ebuf,
                             const float* __restrict__ m,
                             float* __restrict__ denom) {
  const long t = (long)blockIdx.x * blockDim.x + threadIdx.x;
  if (t >= kEtot * kH) return;
  const long e = t >> 2;
  const int  h = (int)(t & 3);
  const long d = (e < kE) ? ei[kE + e] : (e - kE);
  const float ex = __expf(ebuf[t] - m[d * kH + h]);
  ebuf[t] = ex;
  atomicAdd(&denom[d * kH + h], ex);
}

// ------------- pass 3: out[dst] += alpha * xw[src]  (wave per edge) ---------
__global__ __launch_bounds__(256)
void gat_scatter(const long long* __restrict__ ei,
                 const float* __restrict__ ebuf,
                 const float* __restrict__ denom,
                 const float* __restrict__ xw, float* __restrict__ acc) {
  const long e = (long)blockIdx.x * 8 + (threadIdx.x >> 5);
  const int lane = threadIdx.x & 31;
  if (e >= kEtot) return;
  long s, d;
  if (e < kE) { s = ei[e]; d = ei[kE + e]; } else { s = d = e - kE; }
  const int h = lane >> 3;                              // 16 chans per lane
  const float alpha = ebuf[e * kH + h] / (denom[d * kH + h] + 1e-16f);
  const float4* xs = (const float4*)(xw + s * kHD) + lane * 4;
  float* ad = acc + d * kHD + lane * 16;
#pragma unroll
  for (int i = 0; i < 4; ++i) {
    const float4 v = xs[i];
    atomicAdd(ad + 4 * i + 0, alpha * v.x);
    atomicAdd(ad + 4 * i + 1, alpha * v.y);
    atomicAdd(ad + 4 * i + 2, alpha * v.z);
    atomicAdd(ad + 4 * i + 3, alpha * v.w);
  }
}

// --- GEMM2: out[N,128] = elu(acc + bias_gat) @ W_lin[512,128] + b_lin -------
__global__ __launch_bounds__(256)
void gat_gemm2(const float* __restrict__ hacc, const float* __restrict__ bias,
               const float* __restrict__ Wl, const float* __restrict__ bl,
               float* __restrict__ out) {
  __shared__ __align__(16) __bf16 sA[2][BM * BK];
  __shared__ __align__(16) __bf16 sB[2][BN * BK];
  const int tid = threadIdx.x;
  const long row0 = (long)blockIdx.x * BM;
  const int wave = tid >> 5, lane = tid & 31;
  const BiasEluOp op{bias};
  v8f acc[8] = {};

  stage_A(sA[0], hacc, row0, kHD, 0, tid, op);
  stage_B(sB[0], Wl, kD, 0, 0, tid);
  __syncthreads();

  constexpr int nIter = kHD / BK;         // 16
  for (int it = 0; it < nIter; ++it) {
    const int cur = it & 1;
    if (it + 1 < nIter) {                 // stage next tile into alt buffer
      stage_A(sA[cur ^ 1], hacc, row0, kHD, (it + 1) * BK, tid, op);
      stage_B(sB[cur ^ 1], Wl, kD, (it + 1) * BK, 0, tid);
    }
    wmma_tile(sA[cur], sB[cur], acc, lane, wave);
    __syncthreads();
  }

  const int half = lane >> 4, l16 = lane & 15, m0 = wave * 16;
#pragma unroll
  for (int nn = 0; nn < 8; ++nn) {
    const float bv = bl[nn * 16 + l16];
#pragma unroll
    for (int r = 0; r < 8; ++r)
      out[(row0 + m0 + r + 8 * half) * (long)kD + nn * 16 + l16] =
          acc[nn][r] + bv;
  }
}

// ---------------------------------------------------------------------------
extern "C" void kernel_launch(void* const* d_in, const int* in_sizes, int n_in,
                              void* d_out, int out_size, void* d_ws,
                              size_t ws_size, hipStream_t stream) {
  (void)in_sizes; (void)n_in; (void)out_size; (void)ws_size;
  const float*     patches = (const float*)d_in[0];       // [N,128] flat
  const long long* edge_ix = (const long long*)d_in[1];   // int64 [2,E]
  const float*     W_gat   = (const float*)d_in[2];       // [128,512]
  const float*     att_src = (const float*)d_in[3];       // [4,128] flat
  const float*     att_dst = (const float*)d_in[4];
  const float*     bias_g  = (const float*)d_in[5];       // [512]
  const float*     W_lin   = (const float*)d_in[6];       // [512,128]
  const float*     b_lin   = (const float*)d_in[7];       // [128]
  float*           out     = (float*)d_out;               // [N,128]

  // workspace partition (≈ 532 MB total)
  char* p = (char*)d_ws;
  auto take = [&](size_t bytes) {
    void* q = (void*)p;
    p += (bytes + 255) & ~(size_t)255;
    return q;
  };
  float* xw    = (float*)take((size_t)kN * kHD * 4);   // 268 MB
  float* accb  = (float*)take((size_t)kN * kHD * 4);   // 268 MB
  float* a_s   = (float*)take((size_t)kN * kH * 4);
  float* a_d   = (float*)take((size_t)kN * kH * 4);
  float* mmax  = (float*)take((size_t)kN * kH * 4);
  float* denom = (float*)take((size_t)kN * kH * 4);
  float* ebuf  = (float*)take((size_t)kEtot * kH * 4); // 12.6 MB

  // 0) zero accumulator (b128 stores), m=-inf, denom=0
  gat_init<<<(unsigned)((kN * kHD / 4) / 256), 256, 0, stream>>>(
      (float4*)accb, mmax, denom);
  // 1) xw = x @ W_gat   (WMMA bf16, double-buffered)
  gat_gemm1<<<(unsigned)((kN / BM) * (kHD / BN)), 256, 0, stream>>>(patches,
                                                                   W_gat, xw);
  // 2) per-node attention coefficients
  gat_attn<<<(unsigned)(kN / 8), 256, 0, stream>>>(xw, att_src, att_dst, a_s,
                                                   a_d);
  // 3) edge softmax: max, then exp/sum
  const unsigned egrid = (unsigned)((kEtot * kH + 255) / 256);
  gat_edge_logits<<<egrid, 256, 0, stream>>>(edge_ix, a_s, a_d, ebuf, mmax);
  gat_edge_exp<<<egrid, 256, 0, stream>>>(edge_ix, ebuf, mmax, denom);
  // 4) weighted scatter (bandwidth-dominant phase)
  gat_scatter<<<(unsigned)(kEtot / 8), 256, 0, stream>>>(edge_ix, ebuf, denom,
                                                         xw, accb);
  // 5) out = elu(acc + bias) @ W_lin + b_lin   (WMMA bf16, fused activation)
  gat_gemm2<<<(unsigned)(kN / BM), 256, 0, stream>>>(accb, bias_g, W_lin,
                                                     b_lin, out);
}
